// CausalAttention_60997125538304
// MI455X (gfx1250) — matrix operand order repaired and checked by reading, not compiled, not run on hardware
//
#include <hip/hip_runtime.h>

// ---------------------------------------------------------------------------
// Causal attention (B=4, S=2048, D=1024), fp32 in/out, via CDNA5 bf16 WMMA
// (v_wmma_f32_16x16x32_bf16) with double-buffered async global->LDS copies
// (global_load_async_to_lds_b128 / s_wait_asynccnt) on gfx1250 (wave32).
//
// Pipeline (all on `stream`):
//   0) convert x, Wq, Wk, Wv fp32 -> bf16 once (one cheap streaming pass)
//   1) Q = x@Wq^T, K = x@Wk^T           bf16 [8192,1024]
//      Vt = (x@Wv^T)^T                  bf16 [1024,8192] (K-contiguous for PV)
//   2) Sc = (Q K^T)/sqrt(1024)          bf16 [4,2048,2048]; upper blocks skipped
//   3) causal row softmax in place (j>i zeroed so PV runs dense)
//   4) out = P @ V                      fp32 -> d_out
//
// Workspace layout (requires >= 102 MiB):
//   [  0, 16)MiB xb  bf16 [8192,1024]
//   [ 16, 18)MiB Wqb bf16 [1024,1024]
//   [ 18, 20)MiB Wkb
//   [ 20, 22)MiB Wvb
//   [ 22, 38)MiB Q   bf16 [8192,1024]
//   [ 38, 54)MiB K   bf16 [8192,1024]
//   [ 54, 70)MiB Vt  bf16 [1024,8192]
//   [ 70,102)MiB Sc/P bf16 [4,2048,2048]
// ---------------------------------------------------------------------------

typedef __attribute__((ext_vector_type(16))) __bf16         v16bf;
typedef __attribute__((ext_vector_type(8)))  float          v8f;
typedef __attribute__((ext_vector_type(4)))  float          v4f;
typedef __attribute__((ext_vector_type(8)))  unsigned short v8us;
typedef __attribute__((ext_vector_type(16))) unsigned short v16us;

#define BM   128
#define BN   128
#define BK   64
#define LSTR 72   // LDS row stride (bf16 elems): 144 B, multiple of 16 B

__device__ inline unsigned short f2bf(float f) {
  union { float f; unsigned u; } c; c.f = f;
  unsigned u = c.u;
  return (unsigned short)((u + 0x7FFFu + ((u >> 16) & 1u)) >> 16);  // RNE
}
__device__ inline float bf2f(unsigned short h) {
  union { unsigned u; float f; } c; c.u = ((unsigned)h) << 16; return c.f;
}

// Issue async copies for one 128x64 bf16 tile: 256 threads, thread t covers
// row t/2, 32 consecutive K elements (64 B) = 4 x B128, INST_OFFSET applies
// to both the global and LDS side (ISA 15.18).
__device__ inline void issue_tile_async(unsigned short* __restrict__ dst_base,
                                        const unsigned short* __restrict__ g,
                                        int ld, int row0, int k0) {
  const int tid = threadIdx.x;
  const int r = tid >> 1;
  const int c = (tid & 1) << 5;  // 0 or 32 elements
  const unsigned short* src = g + (size_t)(row0 + r) * (size_t)ld + (size_t)(k0 + c);
  const unsigned lds = (unsigned)(size_t)(dst_base + r * LSTR + c);
  asm volatile(
      "global_load_async_to_lds_b128 %0, %1, off\n\t"
      "global_load_async_to_lds_b128 %0, %1, off offset:16\n\t"
      "global_load_async_to_lds_b128 %0, %1, off offset:32\n\t"
      "global_load_async_to_lds_b128 %0, %1, off offset:48"
      :: "v"(lds), "v"(src)
      : "memory");
}

// A-fragment (16x32 bf16, ISA 7.12.2): lanes 0-15 hold K[0:8)+K[16:24),
// lanes 16-31 hold K[8:16)+K[24:32) of row M = lane%16.
__device__ inline v16bf frag_a(const unsigned short* lds, int row, int lane, int koff) {
  const int base = row * LSTR + koff + ((lane & 16) ? 8 : 0);
  v8us q0 = *(const v8us*)(lds + base);
  v8us q1 = *(const v8us*)(lds + base + 16);
  v16us t = __builtin_shufflevector(q0, q1, 0, 1, 2, 3, 4, 5, 6, 7,
                                    8, 9, 10, 11, 12, 13, 14, 15);
  return __builtin_bit_cast(v16bf, t);
}

// B-fragment (32x16 bf16): lane holds column N = lane%16; lanes 0-15 K[0:8)+K[16:24),
// lanes 16-31 K[8:16)+K[24:32) from the [N][K] row-major tile.
__device__ inline v16bf frag_b(const unsigned short* lds, int n, int lane, int koff) {
  const int base = n * LSTR + koff + ((lane & 16) ? 8 : 0);
  v8us q0 = *(const v8us*)(lds + base);
  v8us q1 = *(const v8us*)(lds + base + 16);
  v16us t = __builtin_shufflevector(q0, q1, 0, 1, 2, 3, 4, 5, 6, 7,
                                    8, 9, 10, 11, 12, 13, 14, 15);
  return __builtin_bit_cast(v16bf, t);
}

// C = scale * (A[M,K] x B[N,K]^T), batched over blockIdx.z.
// TC: unsigned short (bf16 ws) or float (d_out). TRANS: store C col-major.
// TRI: skip strictly-upper 128x128 blocks (causal score tiles).
template <typename TC, bool TRANS, bool TRI>
__launch_bounds__(256)
__global__ void gemm_bf16_wmma(const unsigned short* __restrict__ A,
                               const unsigned short* __restrict__ B,
                               TC* __restrict__ C,
                               int M, int N, int K, int lda, int ldb, int ldc,
                               long long bsA, long long bsB, long long bsC,
                               float scale) {
  __shared__ __align__(16) unsigned short sA[2][BM * LSTR];
  __shared__ __align__(16) unsigned short sB[2][BN * LSTR];

  const int bz = blockIdx.z;
  A += (size_t)bz * (size_t)bsA;
  B += (size_t)bz * (size_t)bsB;
  C += (size_t)bz * (size_t)bsC;

  const int bm0 = blockIdx.x * BM;
  const int bn0 = blockIdx.y * BN;
  if constexpr (TRI) {
    if (bn0 > bm0) return;  // strictly above diagonal: all-masked
  }

  const int tid  = threadIdx.x;
  const int lane = tid & 31;
  const int wid  = tid >> 5;          // 8 waves
  const int wr0  = (wid & 3) * 32;    // wave tile: 32 rows x 64 cols
  const int wc0  = (wid >> 2) * 64;

  v8f acc[2][4];
  for (int i = 0; i < 2; ++i)
    for (int j = 0; j < 4; ++j)
      for (int r = 0; r < 8; ++r) acc[i][j][r] = 0.0f;

  const int nk = K / BK;

  // Prologue: stage 0 in flight.
  issue_tile_async(sA[0], A, lda, bm0, 0);
  issue_tile_async(sB[0], B, ldb, bn0, 0);

  int cur = 0;
  for (int ki = 0; ki < nk; ++ki) {
    const int nxt = cur ^ 1;
    if (ki + 1 < nk) {
      // Previous iteration's trailing barrier guarantees nobody still reads
      // buffer `nxt`; stream the next stage while we compute this one.
      issue_tile_async(sA[nxt], A, lda, bm0, (ki + 1) * BK);
      issue_tile_async(sB[nxt], B, ldb, bn0, (ki + 1) * BK);
      // 8 just issued remain; in-order completion => current stage's 8 landed.
      asm volatile("s_wait_asynccnt 0x8" ::: "memory");
    } else {
      asm volatile("s_wait_asynccnt 0x0" ::: "memory");
    }
    __syncthreads();  // all lanes' async data visible to all waves

    const unsigned short* cA = sA[cur];
    const unsigned short* cB = sB[cur];
    const int mrow = wr0 + (lane & 15);
    const int ncol = wc0 + (lane & 15);
#pragma unroll
    for (int kk = 0; kk < 2; ++kk) {  // two 16x16x32 K-chunks per stage
      const int koff = kk * 32;
      v16bf af[2], bfr[4];
#pragma unroll
      for (int mt = 0; mt < 2; ++mt) af[mt] = frag_a(cA, mrow + mt * 16, lane, koff);
#pragma unroll
      for (int nt = 0; nt < 4; ++nt) bfr[nt] = frag_b(cB, ncol + nt * 16, lane, koff);
#pragma unroll
      for (int mt = 0; mt < 2; ++mt)
#pragma unroll
        for (int nt = 0; nt < 4; ++nt)
          acc[mt][nt] = __builtin_amdgcn_wmma_f32_16x16x32_bf16(
              false, af[mt], false, bfr[nt], (short)0, acc[mt][nt], false, false);
    }

    __syncthreads();  // everyone done reading buffer `cur`
    cur = nxt;
  }

  // C/D layout (ISA 7.12.2): VGPR r, lanes 0-15 -> M=r, lanes 16-31 -> M=8+r;
  // N = lane%16.
  const int lhalf = (lane >> 4) * 8;
  const int lcol  = lane & 15;
#pragma unroll
  for (int mt = 0; mt < 2; ++mt)
#pragma unroll
    for (int nt = 0; nt < 4; ++nt)
#pragma unroll
      for (int r = 0; r < 8; ++r) {
        const float v = acc[mt][nt][r] * scale;
        const int row = bm0 + wr0 + mt * 16 + r + lhalf;
        const int col = bn0 + wc0 + nt * 16 + lcol;
        const size_t idx = TRANS ? ((size_t)col * (size_t)ldc + (size_t)row)
                                 : ((size_t)row * (size_t)ldc + (size_t)col);
        if constexpr (sizeof(TC) == 2) {
          ((unsigned short*)C)[idx] = f2bf(v);
        } else {
          ((float*)C)[idx] = v;
        }
      }
}

// One-shot fp32 -> bf16 streaming convert, 8 elements/thread.
__launch_bounds__(256)
__global__ void f32_to_bf16(const float* __restrict__ in,
                            unsigned short* __restrict__ out, int n8) {
  const int i = blockIdx.x * 256 + threadIdx.x;
  if (i >= n8) return;
  const v4f f0 = *(const v4f*)(in + (size_t)i * 8);
  const v4f f1 = *(const v4f*)(in + (size_t)i * 8 + 4);
  v8us o;
  o[0] = f2bf(f0[0]); o[1] = f2bf(f0[1]); o[2] = f2bf(f0[2]); o[3] = f2bf(f0[3]);
  o[4] = f2bf(f1[0]); o[5] = f2bf(f1[1]); o[6] = f2bf(f1[2]); o[7] = f2bf(f1[3]);
  *(v8us*)(out + (size_t)i * 8) = o;
}

// In-place causal softmax over bf16 score rows; zeroes j > i.
__launch_bounds__(256)
__global__ void softmax_rows(unsigned short* __restrict__ Sc) {
  const int S = 2048;
  const int row = blockIdx.x;  // [0, B*S)
  const int b = row >> 11;
  const int i = row & (S - 1);
  unsigned short* p = Sc + ((size_t)b * S + i) * (size_t)S;
  const int L = i + 1;
  const int tid = threadIdx.x;
  __shared__ float red[256];

  float m = -__builtin_inff();
  for (int j = tid; j < L; j += 256) m = fmaxf(m, bf2f(p[j]));
  red[tid] = m;
  __syncthreads();
  for (int s = 128; s > 0; s >>= 1) {
    if (tid < s) red[tid] = fmaxf(red[tid], red[tid + s]);
    __syncthreads();
  }
  m = red[0];
  __syncthreads();

  float sum = 0.0f;
  for (int j = tid; j < L; j += 256) sum += __expf(bf2f(p[j]) - m);
  red[tid] = sum;
  __syncthreads();
  for (int s = 128; s > 0; s >>= 1) {
    if (tid < s) red[tid] += red[tid + s];
    __syncthreads();
  }
  const float inv = 1.0f / red[0];

  for (int j = tid; j < L; j += 256) p[j] = f2bf(__expf(bf2f(p[j]) - m) * inv);
  for (int j = L + tid; j < S; j += 256) p[j] = 0;  // dense PV sees exact zeros
}

extern "C" void kernel_launch(void* const* d_in, const int* in_sizes, int n_in,
                              void* d_out, int out_size, void* d_ws, size_t ws_size,
                              hipStream_t stream) {
  const float* x  = (const float*)d_in[0];   // [4,2048,1024]
  const float* Wq = (const float*)d_in[1];   // [1024,1024]
  const float* Wk = (const float*)d_in[2];
  const float* Wv = (const float*)d_in[3];
  float* out = (float*)d_out;                // [4,2048,1024]

  char* ws = (char*)d_ws;
  unsigned short* xb  = (unsigned short*)(ws);
  unsigned short* Wqb = (unsigned short*)(ws + ((size_t)16 << 20));
  unsigned short* Wkb = (unsigned short*)(ws + ((size_t)18 << 20));
  unsigned short* Wvb = (unsigned short*)(ws + ((size_t)20 << 20));
  unsigned short* Q   = (unsigned short*)(ws + ((size_t)22 << 20));
  unsigned short* Kb  = (unsigned short*)(ws + ((size_t)38 << 20));
  unsigned short* Vt  = (unsigned short*)(ws + ((size_t)54 << 20));
  unsigned short* Sc  = (unsigned short*)(ws + ((size_t)70 << 20));

  const dim3 blk(256, 1, 1);
  const int BSflat = 4 * 2048;        // 8192
  const float inv_sqrt_d = 0.03125f;  // 1/sqrt(1024)

  // 0) fp32 -> bf16 of all GEMM operands (streaming, trivially BW-bound).
  f32_to_bf16<<<dim3(4096, 1, 1), blk, 0, stream>>>(x,  xb,  (8192 * 1024) / 8);
  f32_to_bf16<<<dim3(512, 1, 1),  blk, 0, stream>>>(Wq, Wqb, (1024 * 1024) / 8);
  f32_to_bf16<<<dim3(512, 1, 1),  blk, 0, stream>>>(Wk, Wkb, (1024 * 1024) / 8);
  f32_to_bf16<<<dim3(512, 1, 1),  blk, 0, stream>>>(Wv, Wvb, (1024 * 1024) / 8);

  // 1) QKV projections: [8192,1024] x [1024,1024]^T
  gemm_bf16_wmma<unsigned short, false, false>
      <<<dim3(64, 8, 1), blk, 0, stream>>>(xb, Wqb, Q, BSflat, 1024, 1024,
                                           1024, 1024, 1024, 0, 0, 0, 1.0f);
  gemm_bf16_wmma<unsigned short, false, false>
      <<<dim3(64, 8, 1), blk, 0, stream>>>(xb, Wkb, Kb, BSflat, 1024, 1024,
                                           1024, 1024, 1024, 0, 0, 0, 1.0f);
  gemm_bf16_wmma<unsigned short, true, false>  // Vt[e][b*S+s]
      <<<dim3(64, 8, 1), blk, 0, stream>>>(xb, Wvb, Vt, BSflat, 1024, 1024,
                                           1024, 1024, BSflat, 0, 0, 0, 1.0f);

  // 2) Scores: per batch, Sc = (Q K^T)/32, lower-triangular tiles only.
  gemm_bf16_wmma<unsigned short, false, true>
      <<<dim3(16, 16, 4), blk, 0, stream>>>(Q, Kb, Sc, 2048, 2048, 1024,
                                            1024, 1024, 2048,
                                            (long long)2048 * 1024,
                                            (long long)2048 * 1024,
                                            (long long)2048 * 2048, inv_sqrt_d);

  // 3) Causal softmax in place (one block per row).
  softmax_rows<<<dim3(8192, 1, 1), blk, 0, stream>>>(Sc);

  // 4) out = P @ V : A = P [2048,2048] per batch; B = Vt rows (ld 8192,
  //    batch offset 2048 in the flattened [1024,8192] layout).
  gemm_bf16_wmma<float, false, false>
      <<<dim3(16, 8, 4), blk, 0, stream>>>(Sc, Vt, out, 2048, 1024, 2048,
                                           2048, BSflat, 1024,
                                           (long long)2048 * 2048,
                                           (long long)2048,
                                           (long long)2048 * 1024, 1.0f);
}
